// GarNet_39685497815054
// MI455X (gfx1250) — matrix operand order repaired and checked by reading, not compiled, not run on hardware
//
#include <hip/hip_runtime.h>
#include <hip/hip_bf16.h>
#include <stdint.h>

typedef _Float16 v16h __attribute__((ext_vector_type(16)));
typedef _Float16 v8h  __attribute__((ext_vector_type(8)));
typedef float    v8f  __attribute__((ext_vector_type(8)));

#define B_      64
#define V_      512
#define FIN_    8
#define P_      128
#define A_      16
#define NF_     32
#define F_      144            // P + A
#define TWOF_   288
#define CHUNK_  64
#define NCHUNK_ (V_ / CHUNK_)  // 8
#define EMIN_   0.5f

// LDS budget: 16(ewT) + 18(f) + 18(union agg/dat) + 2(G) = ~54 KB per workgroup
__global__ __launch_bounds__(256) void garnet_cdna5_kernel(
    const float* __restrict__ data,   // (B,V,8)
    const float* __restrict__ W_flr,  // (8,128)
    const float* __restrict__ b_flr,  // (128)
    const float* __restrict__ W_s,    // (8,16)
    const float* __restrict__ b_s,    // (16)
    const float* __restrict__ W_out,  // (4632,32)
    const float* __restrict__ b_out,  // (32)
    float* __restrict__ out)          // (B,V,32)
{
    __shared__ __align__(32) _Float16 s_ewT[A_][V_];     // ew transposed (a, v)     16 KB
    __shared__ __align__(32) _Float16 s_f[F_][CHUNK_];   // f chunk, (feature, vloc) 18 KB
    // dat is only read before the last chunk-loop barrier; agg only written after it.
    __shared__ __align__(16) union SMU {
        float agg[A_][TWOF_];                            // [max(144) | mean(144)]   18 KB
        float dat[V_][FIN_];                             // async-staged data slab   16 KB
    } u_sm;
    __shared__ float s_G[A_][NF_];                       // folded W_out per batch    2 KB

    const int b    = blockIdx.x;
    const int tid  = threadIdx.x;
    const int lane = tid & 31;   // wave32
    const int wave = tid >> 5;   // 8 waves
    const int m    = lane & 15;  // WMMA row/col selector within fragment
    const int g    = lane >> 4;  // half-wave group
    // force scalar (SGPR) branch: WMMA must run with EXEC all-ones, no saveexec
    const bool w0  = (__builtin_amdgcn_readfirstlane(wave) == 0);

    const float* dataB = data + (size_t)b * V_ * FIN_;

    // ---- async-stage the batch's data slab into LDS (16 KB, 4 x b128/thread) ----
    {
        const unsigned lds_base = (unsigned)(uintptr_t)&u_sm.dat[0][0];
        const unsigned long long gbase = (unsigned long long)(uintptr_t)dataB;
        #pragma unroll
        for (int it = 0; it < 4; ++it) {
            const unsigned off = (unsigned)(tid + it * 256) * 16u;
            const unsigned lds_addr = lds_base + off;
            const unsigned long long gaddr = gbase + off;
            asm volatile("global_load_async_to_lds_b128 %0, %1, off"
                         :: "v"(lds_addr), "v"(gaddr) : "memory");
        }
        asm volatile("s_wait_asynccnt 0x0" ::: "memory");
    }
    __syncthreads();

    // ---- phase 0: ew = mask * exp(-dist^2), transposed into LDS ----
    for (int v = tid; v < V_; v += 256) {
        float d[FIN_];
        #pragma unroll
        for (int i = 0; i < FIN_; ++i) d[i] = u_sm.dat[v][i];
        const float mask = (d[3] > EMIN_) ? 1.f : 0.f;
        #pragma unroll
        for (int a = 0; a < A_; ++a) {
            float s = b_s[a];
            #pragma unroll
            for (int i = 0; i < FIN_; ++i) s = fmaf(d[i], W_s[i * A_ + a], s);
            s_ewT[a][v] = (_Float16)(mask * __expf(-s * s));
        }
    }
    __syncthreads();

    // per-wave WMMA accumulators: wave w owns output tile jbase = 16*w;
    // wave 0 additionally owns tile jbase = 128 (9 tiles of 144 features)
    v8f acc0 = {0.f, 0.f, 0.f, 0.f, 0.f, 0.f, 0.f, 0.f};
    v8f acc1 = acc0;

    // running max for agg_max: 9 (a,j) pairs per thread (256*9 == 16*144)
    float mx[9];
    #pragma unroll
    for (int i = 0; i < 9; ++i) mx[i] = -3.402823466e+38f;

    for (int ch = 0; ch < NCHUNK_; ++ch) {
        const int vb = ch * CHUNK_;

        // ---- phase A: build f = [mask*feat | ew] chunk (4 threads / vertex) ----
        {
            const int vl  = tid >> 2;
            const int sub = tid & 3;
            const int v   = vb + vl;
            float d[FIN_];
            #pragma unroll
            for (int i = 0; i < FIN_; ++i) d[i] = u_sm.dat[v][i];
            const float mask = (d[3] > EMIN_) ? 1.f : 0.f;
            #pragma unroll
            for (int pi = 0; pi < 32; ++pi) {
                const int p = sub * 32 + pi;
                float s = b_flr[p];
                #pragma unroll
                for (int i = 0; i < FIN_; ++i) s = fmaf(d[i], W_flr[i * P_ + p], s);
                s_f[p][vl] = (_Float16)(mask * s);
            }
            if (sub == 3) {
                #pragma unroll
                for (int a = 0; a < A_; ++a) s_f[P_ + a][vl] = s_ewT[a][v];
            }
        }
        __syncthreads();

        // ---- phase B1: agg_mean partial sums via v_wmma_f32_16x16x32_f16 ----
        // A(16x32) = ew^T slab, B(32x16) = f-feature tile, C/D f32 16x16
        #pragma unroll
        for (int ks = 0; ks < 2; ++ks) {
            const int lv = ks * 32;
            v16h afrag;
            {
                // A-layout: lane L: M=L%16; K = (L/16)*16 + {0..15} contiguous
                const v8h lo = *(const v8h*)&s_ewT[m][vb + lv + g * 16];
                const v8h hi = *(const v8h*)&s_ewT[m][vb + lv + g * 16 + 8];
                #pragma unroll
                for (int i = 0; i < 8; ++i) { afrag[i] = lo[i]; afrag[8 + i] = hi[i]; }
            }
            {
                // B-layout: lane L: N=L%16; K = (L/16)*16 + {0..15} contiguous
                const int j0 = wave * 16 + m;
                v16h bfrag;
                const v8h lo = *(const v8h*)&s_f[j0][lv + g * 16];
                const v8h hi = *(const v8h*)&s_f[j0][lv + g * 16 + 8];
                #pragma unroll
                for (int i = 0; i < 8; ++i) { bfrag[i] = lo[i]; bfrag[8 + i] = hi[i]; }
                acc0 = __builtin_amdgcn_wmma_f32_16x16x32_f16(
                    false, afrag, false, bfrag, (short)0, acc0, false, false);
            }
            if (w0) {   // scalar-uniform branch: EXEC stays all-ones around WMMA
                const int j0 = P_ + m;
                v16h bfrag;
                const v8h lo = *(const v8h*)&s_f[j0][lv + g * 16];
                const v8h hi = *(const v8h*)&s_f[j0][lv + g * 16 + 8];
                #pragma unroll
                for (int i = 0; i < 8; ++i) { bfrag[i] = lo[i]; bfrag[8 + i] = hi[i]; }
                acc1 = __builtin_amdgcn_wmma_f32_16x16x32_f16(
                    false, afrag, false, bfrag, (short)0, acc1, false, false);
            }
        }

        // ---- phase B2: agg_max partials (register-resident running max) ----
        #pragma unroll
        for (int i = 0; i < 9; ++i) {
            const int pi = tid * 9 + i;
            const int a  = pi / F_;
            const int j  = pi % F_;
            float best = mx[i];
            for (int vl = 0; vl < CHUNK_; ++vl) {
                best = fmaxf(best, (float)s_ewT[a][vb + vl] * (float)s_f[j][vl]);
            }
            mx[i] = best;
        }
        __syncthreads();   // also fences u_sm.dat's last read before agg overwrite
    }

    // ---- write aggregated = [max | mean] into LDS (union: dat is dead now) ----
    {
        const float inv = 1.0f / (float)V_;
        // D-layout: VGPR r holds row M = g*8 + r, col N = m
        #pragma unroll
        for (int r = 0; r < 8; ++r) {
            const int mm = g * 8 + r;
            u_sm.agg[mm][F_ + wave * 16 + m] = acc0[r] * inv;
            if (w0) u_sm.agg[mm][F_ + P_ + m] = acc1[r] * inv;
        }
    }
    #pragma unroll
    for (int i = 0; i < 9; ++i) {
        const int pi = tid * 9 + i;
        u_sm.agg[pi / F_][pi % F_] = mx[i];
    }
    __syncthreads();

    // ---- phase G: fold aggregated through W_out upd-block + direct ew rows ----
    for (int o = tid; o < A_ * NF_; o += 256) {
        const int a  = o / NF_;
        const int nf = o % NF_;
        const float* wrow = W_out + (size_t)(FIN_ + a * TWOF_) * NF_ + nf;
        __builtin_prefetch(wrow, 0, 1);   // global_prefetch_b8
        float s = W_out[(size_t)(FIN_ + 2 * A_ * F_ + a) * NF_ + nf]; // ew direct weight
        for (int j = 0; j < TWOF_; ++j)
            s = fmaf(u_sm.agg[a][j], wrow[(size_t)j * NF_], s);
        s_G[a][nf] = s;
    }
    __syncthreads();

    // ---- final: out = mask * tanh(data@W_data + ew@G + b_out) ----
    for (int v = tid; v < V_; v += 256) {
        float d[FIN_];
        #pragma unroll
        for (int i = 0; i < FIN_; ++i) d[i] = dataB[v * FIN_ + i];
        const float mask = (d[3] > EMIN_) ? 1.f : 0.f;
        float e[A_];
        #pragma unroll
        for (int a = 0; a < A_; ++a) e[a] = (float)s_ewT[a][v];
        float* outp = out + ((size_t)b * V_ + v) * NF_;
        #pragma unroll 4
        for (int nf = 0; nf < NF_; ++nf) {
            float s = b_out[nf];
            #pragma unroll
            for (int i = 0; i < FIN_; ++i) s = fmaf(d[i], W_out[i * NF_ + nf], s);
            #pragma unroll
            for (int a = 0; a < A_; ++a) s = fmaf(e[a], s_G[a][nf], s);
            outp[nf] = mask * tanhf(s);
        }
    }
}

extern "C" void kernel_launch(void* const* d_in, const int* in_sizes, int n_in,
                              void* d_out, int out_size, void* d_ws, size_t ws_size,
                              hipStream_t stream) {
    (void)in_sizes; (void)n_in; (void)out_size; (void)d_ws; (void)ws_size;
    const float* data  = (const float*)d_in[0];
    const float* W_flr = (const float*)d_in[1];
    const float* b_flr = (const float*)d_in[2];
    const float* W_s   = (const float*)d_in[3];
    const float* b_s   = (const float*)d_in[4];
    const float* W_out = (const float*)d_in[5];
    const float* b_out = (const float*)d_in[6];
    float* out = (float*)d_out;
    garnet_cdna5_kernel<<<B_, 256, 0, stream>>>(data, W_flr, b_flr, W_s, b_s,
                                                W_out, b_out, out);
}
